// PointNetPP_1589137899523
// MI455X (gfx1250) — matrix operand order repaired and checked by reading, not compiled, not run on hardware
//
#include <hip/hip_runtime.h>

typedef __attribute__((ext_vector_type(16))) _Float16 v16h;
typedef __attribute__((ext_vector_type(8)))  _Float16 v8h;
typedef __attribute__((ext_vector_type(8)))  float    v8f;

#define NSAMPLE 32

// ---------------------------------------------------------------------------
// Farthest point sampling: one block per batch. dist[] lives in LDS (n<=4096).
// Matches jax.lax.scan semantics: emits current 'far' (starting at 0), then
// computes next via argmax of running min-distance (first-max tie-break).
// ---------------------------------------------------------------------------
__global__ void fps_kernel(const float* __restrict__ xyz, int stride, int n, int npoint,
                           int* __restrict__ fidx, float* __restrict__ centers)
{
    __shared__ float dist[4096];
    __shared__ float rv[256];
    __shared__ int   ri[256];
    __shared__ float cur[3];
    __shared__ int   sfar;
    const int b   = blockIdx.x;
    const int tid = threadIdx.x;
    const float* base = xyz + (size_t)b * n * stride;
    for (int i = tid; i < n; i += 256) dist[i] = 1e10f;
    if (tid == 0) sfar = 0;
    __syncthreads();
    for (int s = 0; s < npoint; ++s) {
        if (tid == 0) {
            int far = sfar;
            float x = base[(size_t)far * stride + 0];
            float y = base[(size_t)far * stride + 1];
            float z = base[(size_t)far * stride + 2];
            cur[0] = x; cur[1] = y; cur[2] = z;
            fidx[(size_t)b * npoint + s] = far;
            float* c = centers + ((size_t)b * npoint + s) * 3;
            c[0] = x; c[1] = y; c[2] = z;
        }
        __syncthreads();
        const float c0 = cur[0], c1 = cur[1], c2 = cur[2];
        float bv = -1.0f; int bi = 0;
        for (int i = tid; i < n; i += 256) {
            float dx = base[(size_t)i * stride + 0] - c0;
            float dy = base[(size_t)i * stride + 1] - c1;
            float dz = base[(size_t)i * stride + 2] - c2;
            float d  = dx * dx + dy * dy + dz * dz;
            float dm = dist[i];
            dm = d < dm ? d : dm;
            dist[i] = dm;
            if (dm > bv) { bv = dm; bi = i; }
        }
        rv[tid] = bv; ri[tid] = bi;
        __syncthreads();
        for (int off = 128; off > 0; off >>= 1) {
            if (tid < off) {
                if (rv[tid + off] > rv[tid] ||
                    (rv[tid + off] == rv[tid] && ri[tid + off] < ri[tid])) {
                    rv[tid] = rv[tid + off];
                    ri[tid] = ri[tid + off];
                }
            }
            __syncthreads();
        }
        if (tid == 0) sfar = ri[0];
        __syncthreads();
    }
}

// ---------------------------------------------------------------------------
// Ball query + group + f16 A-matrix build. One wave per center. Ascending
// index scan with ballot prefix == "sorted smallest nsample in-radius
// indices, padded with the first" (reference semantics). Each lane owns one
// sample row and writes [dx,dy,dz, feats..., 0-pad] as f16.
// ---------------------------------------------------------------------------
__global__ void ballgroup_kernel(const float* __restrict__ xyz, int xstride,
                                 const float* __restrict__ ctr,
                                 const float* __restrict__ feat, int fstride, int cfeat,
                                 int n, int scount, float r2, int cinpad,
                                 _Float16* __restrict__ A)
{
    __shared__ int selbuf[4][NSAMPLE];
    const int lane = threadIdx.x & 31;
    const int wv   = threadIdx.x >> 5;
    const int w    = blockIdx.x * 4 + wv;
    const int b    = w / scount;
    const int s    = w % scount;
    const float* cb = ctr + ((size_t)b * scount + s) * 3;
    const float c0 = cb[0], c1 = cb[1], c2 = cb[2];
    const float* xb = xyz + (size_t)b * n * xstride;
    int fill = 0;
    for (int basei = 0; basei < n; basei += 32) {
        int i = basei + lane;
        bool in = false;
        if (i < n) {
            const float* p = xb + (size_t)i * xstride;
            float dx = p[0] - c0, dy = p[1] - c1, dz = p[2] - c2;
            in = (dx * dx + dy * dy + dz * dz) < r2;
        }
        unsigned mask = (unsigned)__ballot(in);
        if (in) {
            int slot = fill + __popc(mask & ((1u << lane) - 1u));
            if (slot < NSAMPLE) selbuf[wv][slot] = i;
        }
        fill += __popc(mask);
    }
    __syncthreads();
    int idx = (lane < fill) ? selbuf[wv][lane] : selbuf[wv][0];
    const float* p = xb + (size_t)idx * xstride;
    const float* f = feat + ((size_t)b * n + idx) * fstride;
    _Float16* row = A + ((size_t)w * NSAMPLE + lane) * cinpad;
    row[0] = (_Float16)(p[0] - c0);
    row[1] = (_Float16)(p[1] - c1);
    row[2] = (_Float16)(p[2] - c2);
    int j = 3;
    for (int c = 0; c < cfeat; ++c) row[j++] = (_Float16)f[c];
    for (; j < cinpad; ++j) row[j] = (_Float16)0.0f;
}

// ---------------------------------------------------------------------------
// Pack f32 weights [cout][cin] -> f16 transposed [cinpad][cout], zero-padded.
// Each WMMA B-fragment lane reads K rows strided by cout; adjacent lanes read adjacent columns (lane = N).
// ---------------------------------------------------------------------------
__global__ void packw_kernel(const float* __restrict__ W, int cin, int cinpad, int cout,
                             _Float16* __restrict__ Wt)
{
    size_t t = (size_t)blockIdx.x * 256 + threadIdx.x;
    size_t total = (size_t)cinpad * cout;
    if (t >= total) return;
    int k = (int)(t / cout);
    int o = (int)(t % cout);
    float v = (k < cin) ? W[(size_t)o * cin + k] : 0.0f;
    Wt[t] = (_Float16)v;
}

// ---------------------------------------------------------------------------
// Epilogue helper: y = relu(acc*gamma + beta), store f16 column (n fixed per
// lane, m varies per accumulator register).
// ---------------------------------------------------------------------------
__device__ __forceinline__ void store_tile_relu(v8f acc,
                                                const float* __restrict__ gamma,
                                                const float* __restrict__ beta,
                                                _Float16* __restrict__ Out,
                                                int cout, int m0, int hg, int nn)
{
    const float g  = gamma[nn];
    const float bt = beta[nn];
    _Float16* op = Out + (size_t)(m0 + hg * 8) * cout + nn;
#pragma unroll
    for (int r = 0; r < 8; ++r) {
        float y = acc[r] * g + bt;
        y = y > 0.0f ? y : 0.0f;
        op[(size_t)r * cout] = (_Float16)y;
    }
}

// ---------------------------------------------------------------------------
// MLP layer GEMM: Out[R][cout] = relu(A[R][cinpad] * Wt^T * gamma + beta).
// One wave owns a 32x64 output tile: 2 M-tiles x 4 N-tiles = 8 accumulators.
// B-fragments (weights) are reused across both M-tiles -> 24 B loaded per
// WMMA instead of 40, and 8 independent WMMAs per k-step hide load latency.
// A-frag layout per ISA: lane m=lane&15, half-group hg=lane>>4,
//   halves 0..7 -> k = kb + hg*8 + e, halves 8..15 -> k = kb + 16 + hg*8 + e.
// B-frag: lane n=lane&15, halves follow the same k order as the A-frag.
// ---------------------------------------------------------------------------
__global__ void wmma_mlp_kernel(const _Float16* __restrict__ A, int R, int cinpad,
                                const _Float16* __restrict__ Wt, int cout,
                                const float* __restrict__ gamma,
                                const float* __restrict__ beta,
                                _Float16* __restrict__ Out)
{
    const int lane = threadIdx.x & 31;
    const int wv   = threadIdx.x >> 5;
    const int mt   = blockIdx.x * 4 + wv;     // 32-row tiles
    if (mt * 32 >= R) return;
    const int m0 = mt << 5;
    const int n0 = blockIdx.y << 6;
    const int hg = lane >> 4;
    const int ml = lane & 15;
    const _Float16* arow0 = A + (size_t)(m0 + ml) * cinpad + hg * 8;
    const _Float16* arow1 = A + (size_t)(m0 + 16 + ml) * cinpad + hg * 8;
    v8f acc[8];
#pragma unroll
    for (int i = 0; i < 8; ++i) acc[i] = (v8f){};
    for (int kb = 0; kb < cinpad; kb += 32) {
        union { v16h v; v8h h[2]; } a0, a1;
        a0.h[0] = *(const v8h*)(arow0 + kb);
        a0.h[1] = *(const v8h*)(arow0 + kb + 16);
        a1.h[0] = *(const v8h*)(arow1 + kb);
        a1.h[1] = *(const v8h*)(arow1 + kb + 16);
        __builtin_prefetch(arow0 + kb + 64, 0, 1);
        __builtin_prefetch(arow1 + kb + 64, 0, 1);
        const _Float16* brow = Wt + (size_t)(kb + hg * 8) * cout + n0 + ml;
        v16h b0, b1, b2, b3;
#pragma unroll
        for (int e = 0; e < 16; ++e) {
            size_t kk = (size_t)((e & 7) + (e >> 3) * 16) * cout;
            b0[e] = brow[kk];
            b1[e] = brow[kk + 16];
            b2[e] = brow[kk + 32];
            b3[e] = brow[kk + 48];
        }
        acc[0] = __builtin_amdgcn_wmma_f32_16x16x32_f16(false, a0.v, false, b0, (short)0, acc[0], false, false);
        acc[1] = __builtin_amdgcn_wmma_f32_16x16x32_f16(false, a0.v, false, b1, (short)0, acc[1], false, false);
        acc[2] = __builtin_amdgcn_wmma_f32_16x16x32_f16(false, a0.v, false, b2, (short)0, acc[2], false, false);
        acc[3] = __builtin_amdgcn_wmma_f32_16x16x32_f16(false, a0.v, false, b3, (short)0, acc[3], false, false);
        acc[4] = __builtin_amdgcn_wmma_f32_16x16x32_f16(false, a1.v, false, b0, (short)0, acc[4], false, false);
        acc[5] = __builtin_amdgcn_wmma_f32_16x16x32_f16(false, a1.v, false, b1, (short)0, acc[5], false, false);
        acc[6] = __builtin_amdgcn_wmma_f32_16x16x32_f16(false, a1.v, false, b2, (short)0, acc[6], false, false);
        acc[7] = __builtin_amdgcn_wmma_f32_16x16x32_f16(false, a1.v, false, b3, (short)0, acc[7], false, false);
    }
#pragma unroll
    for (int t = 0; t < 4; ++t) {
        store_tile_relu(acc[t],     gamma, beta, Out, cout, m0,      hg, n0 + ml + t * 16);
        store_tile_relu(acc[4 + t], gamma, beta, Out, cout, m0 + 16, hg, n0 + ml + t * 16);
    }
}

// ---------------------------------------------------------------------------
// Max-pool over the 32 samples of each group -> f32 features.
// ---------------------------------------------------------------------------
__global__ void maxpool_kernel(const _Float16* __restrict__ In, int sbs, int cout,
                               float* __restrict__ Out)
{
    size_t t = (size_t)blockIdx.x * 256 + threadIdx.x;
    size_t total = (size_t)sbs * cout;
    if (t >= total) return;
    int bs = (int)(t / cout);
    int c  = (int)(t % cout);
    const _Float16* p = In + (size_t)bs * NSAMPLE * cout + c;
    float m = -1e30f;
#pragma unroll 4
    for (int k = 0; k < NSAMPLE; ++k) {
        float v = (float)p[(size_t)k * cout];
        m = v > m ? v : m;
    }
    Out[t] = m;
}

// ---------------------------------------------------------------------------
// FC prep: Afc[b][c*64+s] = feats3[b][s][c]  (transpose(0,2,1).reshape)
// ---------------------------------------------------------------------------
__global__ void fcprep_kernel(const float* __restrict__ feats, _Float16* __restrict__ Afc)
{
    size_t t = (size_t)blockIdx.x * 256 + threadIdx.x;  // 16*32768 total
    int b = (int)(t >> 15);
    int j = (int)(t & 32767);
    int c = j >> 6;
    int s = j & 63;
    Afc[t] = (_Float16)feats[((size_t)(b * 64 + s)) * 512 + c];
}

// ---------------------------------------------------------------------------
// Final FC: out[16][512] = Afc[16][32768] * Wtfc + bias. M=16 is a single
// WMMA row tile; one block of 4 waves, each wave owns 16x128 (8 accs) so the
// single A-fragment per k-step feeds 8 WMMAs on the K=32768 loop.
// ---------------------------------------------------------------------------
__global__ void wmma_fc_kernel(const _Float16* __restrict__ A,
                               const _Float16* __restrict__ Wt,
                               const float* __restrict__ bias,
                               float* __restrict__ Out)
{
    const int lane = threadIdx.x & 31;
    const int wv   = threadIdx.x >> 5;
    const int n0   = wv << 7;                  // 4 waves x 128 cols = 512
    const int hg   = lane >> 4;
    const int ml   = lane & 15;
    const _Float16* arow = A + (size_t)ml * 32768 + hg * 8;
    v8f acc[8];
#pragma unroll
    for (int i = 0; i < 8; ++i) acc[i] = (v8f){};
    for (int kb = 0; kb < 32768; kb += 32) {
        union { v16h v; v8h h[2]; } ua;
        ua.h[0] = *(const v8h*)(arow + kb);
        ua.h[1] = *(const v8h*)(arow + kb + 16);
        __builtin_prefetch(arow + kb + 64, 0, 1);
        const _Float16* brow = Wt + (size_t)(kb + hg * 8) * 512 + n0 + ml;
#pragma unroll
        for (int t = 0; t < 8; ++t) {
            v16h b;
#pragma unroll
            for (int e = 0; e < 16; ++e) b[e] = brow[(size_t)((e & 7) + (e >> 3) * 16) * 512 + t * 16];
            acc[t] = __builtin_amdgcn_wmma_f32_16x16x32_f16(false, ua.v, false, b, (short)0, acc[t], false, false);
        }
    }
#pragma unroll
    for (int t = 0; t < 8; ++t) {
        int nn = n0 + ml + t * 16;
        float bi = bias[nn];
#pragma unroll
        for (int r = 0; r < 8; ++r) {
            Out[(size_t)(hg * 8 + r) * 512 + nn] = acc[t][r] + bi;
        }
    }
}

// ---------------------------------------------------------------------------
// Host orchestration
// ---------------------------------------------------------------------------
extern "C" void kernel_launch(void* const* d_in, const int* in_sizes, int n_in,
                              void* d_out, int out_size, void* d_ws, size_t ws_size,
                              hipStream_t stream) {
    (void)in_sizes; (void)n_in; (void)out_size; (void)ws_size;

    const float* points = (const float*)d_in[0];
    const float* W[9]; const float* Ga[9]; const float* Be[9];
    int p = 1;
    for (int l = 0; l < 9; ++l) {
        W[l]  = (const float*)d_in[p++];
        Ga[l] = (const float*)d_in[p++];
        Be[l] = (const float*)d_in[p++];
    }
    const float* fcw = (const float*)d_in[p++];
    const float* fcb = (const float*)d_in[p++];

    static const int CIN[9]  = {  6, 64, 64, 131, 128, 128, 259, 256, 256};
    static const int CINP[9] = { 32, 64, 64, 160, 128, 128, 288, 256, 256};
    static const int COUT[9] = { 64, 64,128, 128, 128, 256, 256, 256, 512};

    char* ws = (char*)d_ws;
    size_t off = 0;
    auto alloc = [&](size_t bytes) -> void* {
        void* r = ws + off;
        off += (bytes + 255) & ~(size_t)255;
        return r;
    };
    int*      fidx = (int*)alloc((size_t)16 * 1024 * sizeof(int));
    float*    ctr1 = (float*)alloc((size_t)16 * 1024 * 3 * 4);
    float*    ctr2 = (float*)alloc((size_t)16 * 256 * 3 * 4);
    float*    ctr3 = (float*)alloc((size_t)16 * 64 * 3 * 4);
    float*    fe1  = (float*)alloc((size_t)16 * 1024 * 128 * 4);
    float*    fe2  = (float*)alloc((size_t)16 * 256 * 256 * 4);
    float*    fe3  = (float*)alloc((size_t)16 * 64 * 512 * 4);
    _Float16* Wt[9];
    for (int l = 0; l < 9; ++l) Wt[l] = (_Float16*)alloc((size_t)CINP[l] * COUT[l] * 2);
    _Float16* Wtfc = (_Float16*)alloc((size_t)32768 * 512 * 2);
    _Float16* Afc  = (_Float16*)alloc((size_t)16 * 32768 * 2);
    _Float16* bufA = (_Float16*)alloc((size_t)524288 * 64 * 2);   // 64 MB ping
    _Float16* bufB = (_Float16*)alloc((size_t)524288 * 128 * 2);  // 128 MB pong

    // --- pack all weights to f16 transposed layout ---
    for (int l = 0; l < 9; ++l) {
        size_t tot = (size_t)CINP[l] * COUT[l];
        packw_kernel<<<(unsigned)((tot + 255) / 256), 256, 0, stream>>>(
            W[l], CIN[l], CINP[l], COUT[l], Wt[l]);
    }
    {
        size_t tot = (size_t)32768 * 512;
        packw_kernel<<<(unsigned)((tot + 255) / 256), 256, 0, stream>>>(
            fcw, 32768, 32768, 512, Wtfc);
    }

    const int Bb = 16;

    // ---------------- Level 1 ----------------
    fps_kernel<<<Bb, 256, 0, stream>>>(points, 6, 4096, 1024, fidx, ctr1);
    ballgroup_kernel<<<(Bb * 1024) / 4, 128, 0, stream>>>(
        points, 6, ctr1, points + 3, 6, 3, 4096, 1024, 0.09f, 32, bufA);
    int R = Bb * 1024 * NSAMPLE;  // 524288
    wmma_mlp_kernel<<<dim3(R / 128, COUT[0] / 64), 128, 0, stream>>>(bufA, R, CINP[0], Wt[0], COUT[0], Ga[0], Be[0], bufB);
    wmma_mlp_kernel<<<dim3(R / 128, COUT[1] / 64), 128, 0, stream>>>(bufB, R, CINP[1], Wt[1], COUT[1], Ga[1], Be[1], bufA);
    wmma_mlp_kernel<<<dim3(R / 128, COUT[2] / 64), 128, 0, stream>>>(bufA, R, CINP[2], Wt[2], COUT[2], Ga[2], Be[2], bufB);
    maxpool_kernel<<<(unsigned)(((size_t)Bb * 1024 * 128 + 255) / 256), 256, 0, stream>>>(bufB, Bb * 1024, 128, fe1);

    // ---------------- Level 2 ----------------
    fps_kernel<<<Bb, 256, 0, stream>>>(ctr1, 3, 1024, 256, fidx, ctr2);
    ballgroup_kernel<<<(Bb * 256) / 4, 128, 0, stream>>>(
        ctr1, 3, ctr2, fe1, 128, 128, 1024, 256, 0.36f, 160, bufA);
    R = Bb * 256 * NSAMPLE;  // 131072
    wmma_mlp_kernel<<<dim3(R / 128, COUT[3] / 64), 128, 0, stream>>>(bufA, R, CINP[3], Wt[3], COUT[3], Ga[3], Be[3], bufB);
    wmma_mlp_kernel<<<dim3(R / 128, COUT[4] / 64), 128, 0, stream>>>(bufB, R, CINP[4], Wt[4], COUT[4], Ga[4], Be[4], bufA);
    wmma_mlp_kernel<<<dim3(R / 128, COUT[5] / 64), 128, 0, stream>>>(bufA, R, CINP[5], Wt[5], COUT[5], Ga[5], Be[5], bufB);
    maxpool_kernel<<<(unsigned)(((size_t)Bb * 256 * 256 + 255) / 256), 256, 0, stream>>>(bufB, Bb * 256, 256, fe2);

    // ---------------- Level 3 ----------------
    fps_kernel<<<Bb, 256, 0, stream>>>(ctr2, 3, 256, 64, fidx, ctr3);
    ballgroup_kernel<<<(Bb * 64) / 4, 128, 0, stream>>>(
        ctr2, 3, ctr3, fe2, 256, 256, 256, 64, 1.44f, 288, bufA);
    R = Bb * 64 * NSAMPLE;  // 32768
    wmma_mlp_kernel<<<dim3(R / 128, COUT[6] / 64), 128, 0, stream>>>(bufA, R, CINP[6], Wt[6], COUT[6], Ga[6], Be[6], bufB);
    wmma_mlp_kernel<<<dim3(R / 128, COUT[7] / 64), 128, 0, stream>>>(bufB, R, CINP[7], Wt[7], COUT[7], Ga[7], Be[7], bufA);
    wmma_mlp_kernel<<<dim3(R / 128, COUT[8] / 64), 128, 0, stream>>>(bufA, R, CINP[8], Wt[8], COUT[8], Ga[8], Be[8], bufB);
    maxpool_kernel<<<(unsigned)(((size_t)Bb * 64 * 512 + 255) / 256), 256, 0, stream>>>(bufB, Bb * 64, 512, fe3);

    // ---------------- FC head ----------------
    fcprep_kernel<<<(16 * 32768) / 256, 256, 0, stream>>>(fe3, Afc);
    wmma_fc_kernel<<<1, 128, 0, stream>>>(Afc, Wtfc, fcb, (float*)d_out);
}
